// ARCAttention_74277164417694
// MI455X (gfx1250) — matrix operand order repaired and checked by reading, not compiled, not run on hardware
//
#include <hip/hip_runtime.h>

// ---------------------------------------------------------------------------
// Types for CDNA5 WMMA (wave32): v_wmma_f32_16x16x32_bf16
// ---------------------------------------------------------------------------
typedef __attribute__((ext_vector_type(16))) __bf16 bf16x16;
typedef __attribute__((ext_vector_type(8)))  __bf16 bf16x8;
typedef __attribute__((ext_vector_type(8)))  float  f32x8;
typedef __attribute__((ext_vector_type(4)))  unsigned int u32x4;
typedef __attribute__((ext_vector_type(8)))  int i32x8;
typedef __attribute__((ext_vector_type(4)))  int i32x4;

#if __has_builtin(__builtin_amdgcn_tensor_load_to_lds)
#define ATTN_TDM 1
#else
#define ATTN_TDM 0
#endif

__device__ __forceinline__ f32x8 f32x8_zero() {
    f32x8 z = {0.f, 0.f, 0.f, 0.f, 0.f, 0.f, 0.f, 0.f};
    return z;
}

__device__ __forceinline__ __bf16 to_bf16(float f) {
    // round-to-nearest-even f32 -> bf16 via bit manipulation
    unsigned u = __float_as_uint(f);
    unsigned r = u + 0x7FFFu + ((u >> 16) & 1u);
    unsigned short h = (unsigned short)(r >> 16);
    __bf16 out;
    __builtin_memcpy(&out, &h, 2);
    return out;
}

__device__ __forceinline__ bf16x16 cat8(bf16x8 lo, bf16x8 hi) {
    return __builtin_shufflevector(lo, hi, 0,1,2,3,4,5,6,7,8,9,10,11,12,13,14,15);
}

// A-matrix fragment (16x32 bf16). ISA 7.12.2: lanes 0-15 (row M=lane) hold
// K={0..7,16..23}; lanes 16-31 (row M=lane-16) hold K={8..15,24..31}.
__device__ __forceinline__ bf16x16 frag_a(const __bf16* rowptr, int kk, int lane) {
    int kg = (lane >> 4) << 3;                    // 0 or 8
    bf16x8 lo = *(const bf16x8*)(rowptr + kk + kg);
    bf16x8 hi = *(const bf16x8*)(rowptr + kk + kg + 16);
    return cat8(lo, hi);
}

// B-matrix fragment (32x16 bf16). Lane holds column N=lane&15; lanes 0-15
// hold K={0..7,16..23}, lanes 16-31 hold K={8..15,24..31}.
__device__ __forceinline__ bf16x16 frag_b(const __bf16* rowptr, int kk, int lane) {
    int kb = (lane >> 4) << 3;                    // 0 or 8
    bf16x8 lo = *(const bf16x8*)(rowptr + kk + kb);
    bf16x8 hi = *(const bf16x8*)(rowptr + kk + kb + 16);
    return cat8(lo, hi);
}

__device__ __forceinline__ f32x8 wmma_bf16f32(bf16x16 a, bf16x16 b, f32x8 c) {
    return __builtin_amdgcn_wmma_f32_16x16x32_bf16(
        /*neg_a=*/false, a, /*neg_b=*/false, b,
        /*c_mod=*/(short)0, c, /*reuse_a=*/false, /*reuse_b=*/false);
}

// ---------------------------------------------------------------------------
// Tensor Data Mover: async 2D tile load Global -> LDS (ISA ch.8 D# layout).
// Issued per-wave; completion tracked with TENSORcnt.
// ---------------------------------------------------------------------------
__device__ __forceinline__ void tdm_load_2d(const void* gptr, void* ldsptr,
                                            unsigned rowElems, unsigned rows,
                                            unsigned strideElems) {
#if ATTN_TDM
    unsigned long long ga = (unsigned long long)gptr;
    unsigned lds = (unsigned)(unsigned long long)ldsptr;  // low 32b = LDS byte addr
    u32x4 g0;
    g0[0] = 1u;                                  // count=1, user descriptor
    g0[1] = lds;                                 // lds_addr
    g0[2] = (unsigned)ga;                        // global_addr[31:0]
    g0[3] = (unsigned)(ga >> 32) | (2u << 30);   // global_addr[56:32] | type=2
    i32x8 g1;
    g1[0] = (int)(1u << 16);                     // wg_mask=0, data_size=1 (2B)
    g1[1] = (int)((rowElems & 0xFFFFu) << 16);   // tensor_dim0[15:0]
    g1[2] = (int)((rowElems >> 16) | ((rows & 0xFFFFu) << 16)); // td0 hi | td1 lo
    g1[3] = (int)((rows >> 16) | ((rowElems & 0xFFFFu) << 16)); // td1 hi | tile_dim0
    g1[4] = (int)(rows & 0xFFFFu);               // tile_dim1 | tile_dim2=0
    g1[5] = (int)strideElems;                    // tensor_dim0_stride[31:0]
    g1[6] = 0;                                   // stride hi | dim1_stride lo
    g1[7] = 0;
    i32x4 z4 = {0, 0, 0, 0};
#if __clang_major__ >= 23
    i32x8 z8 = {0, 0, 0, 0, 0, 0, 0, 0};
    __builtin_amdgcn_tensor_load_to_lds(g0, g1, z4, z4, z8, 0);
#else
    __builtin_amdgcn_tensor_load_to_lds(g0, g1, z4, z4, 0);
#endif
#else
    (void)gptr; (void)ldsptr; (void)rowElems; (void)rows; (void)strideElems;
#endif
}

__device__ __forceinline__ void tdm_wait() {
#if ATTN_TDM && __has_builtin(__builtin_amdgcn_s_wait_tensorcnt)
    __builtin_amdgcn_s_wait_tensorcnt(0);
#endif
}

// ---------------------------------------------------------------------------
// f32 -> bf16 conversion (grid-stride)
// ---------------------------------------------------------------------------
__global__ __launch_bounds__(256)
void cvt_f32_bf16(const float* __restrict__ in, __bf16* __restrict__ out, long long n) {
    long long stride = (long long)gridDim.x * blockDim.x;
    for (long long i = (long long)blockIdx.x * blockDim.x + threadIdx.x; i < n; i += stride)
        out[i] = to_bf16(in[i]);
}

// ---------------------------------------------------------------------------
// GEMM:  C[M,N] f32 = A[M,K] bf16 @ W[N,K]^T bf16     (row-major)
// Block tile 128x128, K-step 32, 8 waves in a 4(M)x2(N) grid; each wave owns
// a 32x64 patch of C: 2 A-frags x 4 B-frags -> 8 v_wmma per K-step.
// Requires M%128==0, N%128==0, K%32==0 (all shapes comply).
// ---------------------------------------------------------------------------
__global__ __launch_bounds__(256)
void gemm_nt_bf16(const __bf16* __restrict__ A, const __bf16* __restrict__ W,
                  float* __restrict__ C, int M, int N, int K) {
    __shared__ __bf16 sA[128][32];   // 8 KB
    __shared__ __bf16 sB[128][32];   // 8 KB

    const int tid  = threadIdx.x;
    const int lane = tid & 31;
    const int wave = tid >> 5;
    const int waveM = wave >> 1;     // 0..3  (32 rows each)
    const int waveN = wave & 1;      // 0..1  (64 cols each)
    const int bm = blockIdx.y * 128;
    const int bn = blockIdx.x * 128;

    f32x8 acc[2][4];
#pragma unroll
    for (int mi = 0; mi < 2; ++mi)
#pragma unroll
        for (int ni = 0; ni < 4; ++ni) acc[mi][ni] = f32x8_zero();

    for (int k0 = 0; k0 < K; k0 += 32) {
#pragma unroll
        for (int c = 0; c < 2; ++c) {
            int lin = tid + c * 256;           // 0..511
            int row = lin >> 2;                // 0..127
            int col = (lin & 3) * 8;           // 0,8,16,24
            *(bf16x8*)&sA[row][col] =
                *(const bf16x8*)&A[(size_t)(bm + row) * K + k0 + col];
            *(bf16x8*)&sB[row][col] =
                *(const bf16x8*)&W[(size_t)(bn + row) * K + k0 + col];
        }
        __syncthreads();

        bf16x16 af[2], bfr[4];
#pragma unroll
        for (int mi = 0; mi < 2; ++mi)
            af[mi] = frag_a(&sA[waveM * 32 + mi * 16 + (lane & 15)][0], 0, lane);
#pragma unroll
        for (int ni = 0; ni < 4; ++ni)
            bfr[ni] = frag_b(&sB[waveN * 64 + ni * 16 + (lane & 15)][0], 0, lane);
#pragma unroll
        for (int mi = 0; mi < 2; ++mi)
#pragma unroll
            for (int ni = 0; ni < 4; ++ni)
                acc[mi][ni] = wmma_bf16f32(af[mi], bfr[ni], acc[mi][ni]);
        __syncthreads();
    }

    const int cRow = (lane >> 4) * 8;
    const int cCol = lane & 15;
#pragma unroll
    for (int mi = 0; mi < 2; ++mi) {
        float* base = C + (size_t)(bm + waveM * 32 + mi * 16 + cRow) * N +
                      (bn + waveN * 64 + cCol);
#pragma unroll
        for (int ni = 0; ni < 4; ++ni)
#pragma unroll
            for (int r = 0; r < 8; ++r)
                base[(size_t)r * N + ni * 16] = acc[mi][ni][r];
    }
}

// ---------------------------------------------------------------------------
// RMSNorm over first C cols of rows with leading dim ld -> bf16 rows of C.
// ---------------------------------------------------------------------------
__global__ __launch_bounds__(256)
void rmsnorm_kernel(const float* __restrict__ in, int ld,
                    const float* __restrict__ w,
                    __bf16* __restrict__ out, int C) {
    const int row = blockIdx.x;
    const int tid = threadIdx.x;
    const float* x = in + (size_t)row * ld;
    __shared__ float red[256];
    float s = 0.f;
    for (int i = tid; i < C; i += 256) { float v = x[i]; s += v * v; }
    red[tid] = s;
    __syncthreads();
    for (int o = 128; o > 0; o >>= 1) {
        if (tid < o) red[tid] += red[tid + o];
        __syncthreads();
    }
    float inv = rsqrtf(red[0] / (float)C + 1e-6f);
    for (int i = tid; i < C; i += 256)
        out[(size_t)row * C + i] = to_bf16(x[i] * inv * w[i]);
}

// ---------------------------------------------------------------------------
// q epilogue: q f32 (T x 3072 = [h][192]) -> qf bf16 [b*16+h][s][192]
// with RoPE on dims 128..191.  One block per (token, head).
// ---------------------------------------------------------------------------
__global__ void rope_q_kernel(const float* __restrict__ q, __bf16* __restrict__ qf, int S) {
    const int gb = blockIdx.x;
    const int h = gb & 15;
    const int t = gb >> 4;
    const int b = t / S;
    const int s = t % S;
    const int d = threadIdx.x;        // 0..191
    float val = q[(size_t)t * 3072 + h * 192 + d];
    if (d >= 128) {
        int j = d - 128;
        int jj = j & 31;
        float freq = (float)s * __powf(10000.f, -(float)jj / 32.f);
        float c, sn;
        __sincosf(freq, &sn, &c);
        float other = q[(size_t)t * 3072 + h * 192 + 128 + ((j < 32) ? j + 32 : j - 32)];
        val = (j < 32) ? (val * c - other * sn) : (val * c + other * sn);
    }
    qf[(((size_t)(b * 16 + h) * S + s) * 192) + d] = to_bf16(val);
}

// ---------------------------------------------------------------------------
// k/v epilogue -> kf bf16 [bh][s][192] (RoPE on 128..191), vh bf16 [bh][s][128]
// ---------------------------------------------------------------------------
__global__ __launch_bounds__(256)
void kv_epilogue_kernel(const float* __restrict__ kv, const float* __restrict__ kva,
                        __bf16* __restrict__ kf, __bf16* __restrict__ vh, int S) {
    const int gb = blockIdx.x;
    const int h = gb & 15;
    const int t = gb >> 4;
    const int b = t / S;
    const int s = t % S;
    const int d = threadIdx.x;
    const size_t hd = (size_t)(b * 16 + h) * S + s;
    if (d < 128) {
        kf[hd * 192 + d] = to_bf16(kv[(size_t)t * 4096 + h * 256 + d]);
        vh[hd * 128 + d] = to_bf16(kv[(size_t)t * 4096 + h * 256 + 128 + d]);
    } else if (d < 192) {
        int j = d - 128;
        int jj = j & 31;
        float freq = (float)s * __powf(10000.f, -(float)jj / 32.f);
        float c, sn;
        __sincosf(freq, &sn, &c);
        float x = kva[(size_t)t * 1536 + 512 + h * 64 + j];
        float other = kva[(size_t)t * 1536 + 512 + h * 64 + ((j < 32) ? j + 32 : j - 32)];
        float val = (j < 32) ? (x * c - other * sn) : (x * c + other * sn);
        kf[hd * 192 + d] = to_bf16(val);
    }
}

// ---------------------------------------------------------------------------
// Pattern head split: in f32 (T x 2048) -> out bf16 [b*16+h][s][128]
// ---------------------------------------------------------------------------
__global__ __launch_bounds__(256)
void split_heads_kernel(const float* __restrict__ in, __bf16* __restrict__ out, int S) {
    long long total = (long long)2048 * 2048;
    long long stride = (long long)gridDim.x * blockDim.x;
    for (long long i = (long long)blockIdx.x * blockDim.x + threadIdx.x; i < total; i += stride) {
        long long t = i >> 11;
        int c = (int)(i & 2047);
        int h = c >> 7, d = c & 127;
        int b = (int)(t / S), s = (int)(t % S);
        out[(((size_t)(b * 16 + h) * S) + s) * 128 + d] = to_bf16(in[i]);
    }
}

// ---------------------------------------------------------------------------
// Fused flash attention (per (b,h), 64-query tiles, 32-key steps).
// Q,K: [bh][S][D] bf16; V: [bh][S][128] bf16; ctx: token-major (T x H*128) bf16.
// Q/K tiles staged via Tensor Data Mover when available.
// ---------------------------------------------------------------------------
template <int D>
__global__ __launch_bounds__(256)
void attn_kernel(const __bf16* __restrict__ Qh, const __bf16* __restrict__ Kh,
                 const __bf16* __restrict__ Vh, __bf16* __restrict__ Ctx,
                 int S, int H, float scale) {
    const int bh = blockIdx.y;
    const int h = bh % H;
    const int b = bh / H;
    const int q0 = blockIdx.x * 64;
    const int tid = threadIdx.x;
    const int lane = tid & 31;
    const int wave = tid >> 5;

    __shared__ __bf16 sQ[64][D];
    __shared__ __bf16 sK[32][D];
    __shared__ __bf16 sVt[128][32];   // transposed [d][key]
    __shared__ float  sS[64][32];
    __shared__ __bf16 sP[64][32];
    __shared__ float  rowM[64], rowL[64], rowA[64];

    const size_t baseQ = ((size_t)bh * S + q0) * D;
#if ATTN_TDM
    if (wave == 0)
        tdm_load_2d(Qh + baseQ, (void*)&sQ[0][0], D, 64, D);
#else
    for (int i = tid * 8; i < 64 * D; i += 2048)
        *(bf16x8*)((__bf16*)sQ + i) = *(const bf16x8*)(Qh + baseQ + i);
#endif

    if (tid < 64) { rowM[tid] = -1e30f; rowL[tid] = 0.f; }

    f32x8 oacc[4];
#pragma unroll
    for (int i = 0; i < 4; ++i) oacc[i] = f32x8_zero();

    const int mtile = wave >> 1;            // 0..3 : q-row tile
    const int sntile = wave & 1;            // 0..1 : score col tile
    const int ocol0 = (wave & 1) * 64;      // O column block

    const int nsteps = S / 32;
    for (int kb = 0; kb < nsteps; ++kb) {
        // --- stage K tile (TDM when available) and transposed V tile ---
        const size_t baseK = ((size_t)bh * S + kb * 32) * D;
#if ATTN_TDM
        if (wave == 0) {
            tdm_load_2d(Kh + baseK, (void*)&sK[0][0], D, 32, D);
            tdm_wait();                     // covers Q tile on first iteration
        }
#else
        for (int i = tid * 8; i < 32 * D; i += 2048)
            *(bf16x8*)((__bf16*)sK + i) = *(const bf16x8*)(Kh + baseK + i);
#endif
        const size_t baseV = ((size_t)bh * S + kb * 32) * 128;
        for (int i = tid; i < 32 * 128; i += 256) {
            int kr = i >> 7, d = i & 127;
            sVt[d][kr] = Vh[baseV + i];
        }
        __syncthreads();   // (A)

        // --- S = scale * Q K^T (64x32), one 16x16 tile per wave ---
        {
            f32x8 sacc = f32x8_zero();
#pragma unroll
            for (int kk = 0; kk < D; kk += 32) {
                bf16x16 af = frag_a(&sQ[mtile * 16 + (lane & 15)][0], kk, lane);
                bf16x16 bf = frag_b(&sK[sntile * 16 + (lane & 15)][0], kk, lane);
                sacc = wmma_bf16f32(af, bf, sacc);
            }
            int srow = mtile * 16 + (lane >> 4) * 8;
            int scol = sntile * 16 + (lane & 15);
#pragma unroll
            for (int r = 0; r < 8; ++r)
                sS[srow + r][scol] = sacc[r] * scale;
        }
        __syncthreads();   // (B)

        // --- online softmax per row ---
        if (tid < 64) {
            float mold = rowM[tid];
            float mnew = mold;
#pragma unroll 4
            for (int j = 0; j < 32; ++j) mnew = fmaxf(mnew, sS[tid][j]);
            float alpha = __expf(mold - mnew);
            float sum = 0.f;
#pragma unroll 4
            for (int j = 0; j < 32; ++j) {
                float p = __expf(sS[tid][j] - mnew);
                sP[tid][j] = to_bf16(p);
                sum += p;
            }
            rowL[tid] = rowL[tid] * alpha + sum;
            rowM[tid] = mnew;
            rowA[tid] = alpha;
        }
        __syncthreads();   // (C)

        // --- rescale accumulators, then O += P V ---
        {
            int orow = mtile * 16 + (lane >> 4) * 8;
#pragma unroll
            for (int r = 0; r < 8; ++r) {
                float al = rowA[orow + r];
#pragma unroll
                for (int ni = 0; ni < 4; ++ni) oacc[ni][r] *= al;
            }
            bf16x16 pf = frag_a(&sP[mtile * 16 + (lane & 15)][0], 0, lane);
#pragma unroll
            for (int ni = 0; ni < 4; ++ni) {
                bf16x16 vf = frag_b(&sVt[ocol0 + ni * 16 + (lane & 15)][0], 0, lane);
                oacc[ni] = wmma_bf16f32(pf, vf, oacc[ni]);
            }
        }
        __syncthreads();   // (D)
    }

    // --- normalize and store ctx[token][h*128 + d] ---
    {
        int orow = mtile * 16 + (lane >> 4) * 8;
#pragma unroll
        for (int r = 0; r < 8; ++r) {
            float invL = 1.f / rowL[orow + r];
            size_t rowbase = ((size_t)(b * S + q0 + orow + r)) * (size_t)(H * 128) + (size_t)h * 128;
#pragma unroll
            for (int ni = 0; ni < 4; ++ni)
                Ctx[rowbase + ocol0 + ni * 16 + (lane & 15)] = to_bf16(oacc[ni][r] * invL);
        }
    }
}

// ---------------------------------------------------------------------------
// Gate + blend:  gates = softmax(hid @ gate_w^T + gate_b)
// ---------------------------------------------------------------------------
__global__ __launch_bounds__(256)
void gate_blend_kernel(const float* __restrict__ hid, const float* __restrict__ gw,
                       const float* __restrict__ gb, const float* __restrict__ mo,
                       const float* __restrict__ po, float* __restrict__ out) {
    const int t = blockIdx.x;
    const int tid = threadIdx.x;
    const float* x = hid + (size_t)t * 2048;
    __shared__ float r0[256], r1[256];
    float s0 = 0.f, s1 = 0.f;
    for (int i = tid; i < 2048; i += 256) {
        float v = x[i];
        s0 += v * gw[i];
        s1 += v * gw[2048 + i];
    }
    r0[tid] = s0; r1[tid] = s1;
    __syncthreads();
    for (int o = 128; o > 0; o >>= 1) {
        if (tid < o) { r0[tid] += r0[tid + o]; r1[tid] += r1[tid + o]; }
        __syncthreads();
    }
    float g0 = r0[0] + gb[0], g1 = r1[0] + gb[1];
    float m = fmaxf(g0, g1);
    float e0 = __expf(g0 - m), e1 = __expf(g1 - m);
    float inv = 1.f / (e0 + e1);
    float p0 = e0 * inv, p1 = e1 * inv;
    for (int i = tid; i < 2048; i += 256) {
        size_t idx = (size_t)t * 2048 + i;
        out[idx] = p0 * mo[idx] + p1 * po[idx];
    }
}

// ---------------------------------------------------------------------------
// Host orchestration
// ---------------------------------------------------------------------------
extern "C" void kernel_launch(void* const* d_in, const int* in_sizes, int n_in,
                              void* d_out, int out_size, void* d_ws, size_t ws_size,
                              hipStream_t stream) {
    (void)in_sizes; (void)n_in; (void)out_size; (void)ws_size;
    const int Bt = 2, S = 1024, T = 2048, H = 16;

    const float* hid     = (const float*)d_in[0];
    const float* q_a_w   = (const float*)d_in[1];
    const float* q_a_ln  = (const float*)d_in[2];
    const float* q_b_w   = (const float*)d_in[3];
    const float* kv_a_w  = (const float*)d_in[4];
    const float* kv_a_ln = (const float*)d_in[5];
    const float* kv_b_w  = (const float*)d_in[6];
    const float* o_w     = (const float*)d_in[7];
    const float* sp_q_w  = (const float*)d_in[8];
    const float* sp_k_w  = (const float*)d_in[9];
    const float* sp_v_w  = (const float*)d_in[10];
    const float* sp_o_w  = (const float*)d_in[11];
    const float* gate_w  = (const float*)d_in[12];
    const float* gate_b  = (const float*)d_in[13];
    float* out = (float*)d_out;
    char* ws = (char*)d_ws;

    size_t off = 0;
    auto carve = [&](size_t bytes) -> char* {
        char* p = ws + off;
        off += (bytes + 255) & ~(size_t)255;
        return p;
    };

    __bf16* wqa  = (__bf16*)carve((size_t)1536 * 2048 * 2);
    __bf16* wqb  = (__bf16*)carve((size_t)3072 * 1536 * 2);
    __bf16* wkva = (__bf16*)carve((size_t)1536 * 2048 * 2);
    __bf16* wkvb = (__bf16*)carve((size_t)4096 * 512 * 2);
    __bf16* wo   = (__bf16*)carve((size_t)2048 * 2048 * 2);
    __bf16* wspq = (__bf16*)carve((size_t)2048 * 2048 * 2);
    __bf16* wspk = (__bf16*)carve((size_t)2048 * 2048 * 2);
    __bf16* wspv = (__bf16*)carve((size_t)2048 * 2048 * 2);
    __bf16* wspo = (__bf16*)carve((size_t)2048 * 2048 * 2);
    __bf16* hidbf   = (__bf16*)carve((size_t)T * 2048 * 2);
    float*  scratch = (float*)carve((size_t)T * 4096 * 4);
    float*  kva_f32 = (float*)carve((size_t)T * 1536 * 4);
    __bf16* qmid    = (__bf16*)carve((size_t)T * 1536 * 2);
    __bf16* kvl     = (__bf16*)carve((size_t)T * 512 * 2);
    __bf16* qf      = (__bf16*)carve((size_t)Bt * H * S * 192 * 2);
    __bf16* kf      = (__bf16*)carve((size_t)Bt * H * S * 192 * 2);
    __bf16* vh      = (__bf16*)carve((size_t)Bt * H * S * 128 * 2);
    __bf16* ctx     = (__bf16*)carve((size_t)T * 2048 * 2);
    float*  main_out    = (float*)carve((size_t)T * 2048 * 4);
    float*  pattern_out = (float*)carve((size_t)T * 2048 * 4);
    __bf16* pqh  = qf;   // pattern path reuses main-path buffers
    __bf16* pkh  = kf;
    __bf16* pvh  = vh;
    __bf16* pctx = ctx;

    // ---- convert weights + hidden to bf16 ----
    cvt_f32_bf16<<<1024, 256, 0, stream>>>(q_a_w,  wqa,  (long long)1536 * 2048);
    cvt_f32_bf16<<<1024, 256, 0, stream>>>(q_b_w,  wqb,  (long long)3072 * 1536);
    cvt_f32_bf16<<<1024, 256, 0, stream>>>(kv_a_w, wkva, (long long)1536 * 2048);
    cvt_f32_bf16<<<1024, 256, 0, stream>>>(kv_b_w, wkvb, (long long)4096 * 512);
    cvt_f32_bf16<<<1024, 256, 0, stream>>>(o_w,    wo,   (long long)2048 * 2048);
    cvt_f32_bf16<<<1024, 256, 0, stream>>>(sp_q_w, wspq, (long long)2048 * 2048);
    cvt_f32_bf16<<<1024, 256, 0, stream>>>(sp_k_w, wspk, (long long)2048 * 2048);
    cvt_f32_bf16<<<1024, 256, 0, stream>>>(sp_v_w, wspv, (long long)2048 * 2048);
    cvt_f32_bf16<<<1024, 256, 0, stream>>>(sp_o_w, wspo, (long long)2048 * 2048);
    cvt_f32_bf16<<<1024, 256, 0, stream>>>(hid,    hidbf, (long long)T * 2048);

    // ---- main path ----
    gemm_nt_bf16<<<dim3(1536 / 128, T / 128), 256, 0, stream>>>(hidbf, wqa, scratch, T, 1536, 2048);
    rmsnorm_kernel<<<T, 256, 0, stream>>>(scratch, 1536, q_a_ln, qmid, 1536);
    gemm_nt_bf16<<<dim3(3072 / 128, T / 128), 256, 0, stream>>>(qmid, wqb, scratch, T, 3072, 1536);
    rope_q_kernel<<<T * H, 192, 0, stream>>>(scratch, qf, S);
    gemm_nt_bf16<<<dim3(1536 / 128, T / 128), 256, 0, stream>>>(hidbf, wkva, kva_f32, T, 1536, 2048);
    rmsnorm_kernel<<<T, 256, 0, stream>>>(kva_f32, 1536, kv_a_ln, kvl, 512);
    gemm_nt_bf16<<<dim3(4096 / 128, T / 128), 256, 0, stream>>>(kvl, wkvb, scratch, T, 4096, 512);
    kv_epilogue_kernel<<<T * H, 256, 0, stream>>>(scratch, kva_f32, kf, vh, S);
    attn_kernel<192><<<dim3(S / 64, Bt * H), 256, 0, stream>>>(
        qf, kf, vh, ctx, S, H, 0.07216878364870322f /* 1/sqrt(192) */);
    gemm_nt_bf16<<<dim3(2048 / 128, T / 128), 256, 0, stream>>>(ctx, wo, main_out, T, 2048, 2048);

    // ---- pattern path ----
    gemm_nt_bf16<<<dim3(2048 / 128, T / 128), 256, 0, stream>>>(hidbf, wspq, scratch, T, 2048, 2048);
    split_heads_kernel<<<2048, 256, 0, stream>>>(scratch, pqh, S);
    gemm_nt_bf16<<<dim3(2048 / 128, T / 128), 256, 0, stream>>>(hidbf, wspk, scratch, T, 2048, 2048);
    split_heads_kernel<<<2048, 256, 0, stream>>>(scratch, pkh, S);
    gemm_nt_bf16<<<dim3(2048 / 128, T / 128), 256, 0, stream>>>(hidbf, wspv, scratch, T, 2048, 2048);
    split_heads_kernel<<<2048, 256, 0, stream>>>(scratch, pvh, S);
    attn_kernel<128><<<dim3(S / 64, Bt * H), 256, 0, stream>>>(
        pqh, pkh, pvh, pctx, S, H, 0.08838834764831845f /* 1/sqrt(128) */);
    gemm_nt_bf16<<<dim3(2048 / 128, T / 128), 256, 0, stream>>>(pctx, wspo, pattern_out, T, 2048, 2048);

    // ---- gate + blend ----
    gate_blend_kernel<<<T, 256, 0, stream>>>(hid, gate_w, gate_b, main_out, pattern_out, out);
}
